// Agent_89936615178856
// MI455X (gfx1250) — matrix operand order repaired and checked by reading, not compiled, not run on hardware
//
#include <hip/hip_runtime.h>
#include <hip/hip_bf16.h>
#include <math.h>

// ---------------------------------------------------------------------------
// CDNA5 (gfx1250) implementation of the GNN "Agent" reference.
// conv2 (~92% of FLOPs) and conv3 (~6%) run on v_wmma_f32_16x16x32_f16.
// Weights are pre-packed into the ISA 16x32 f16 A-fragment lane layout (one
// 32B L2-resident load per fragment). Activations tile through per-wave LDS in
// f16; B fragments are gathered branch-free via zero-padded im2col LUTs.
// ---------------------------------------------------------------------------

typedef __attribute__((ext_vector_type(16))) _Float16      v16h;
typedef __attribute__((ext_vector_type(8)))  float         v8f;
typedef __attribute__((ext_vector_type(8)))  unsigned short u16x8;

#define NODE_F   64      // P*B = 4*16
#define HID      64
#define GDIM     50
#define H1_SIZE  1440    // 10 * 9 * 16
#define H1_PAD   96      // zero pad: OOB LUT entries land here (noff <= 87)
#define H2_ROWS  50
#define H2_COLS  48      // 6 * 8
#define H2_PAD   32      // slack for the discarded n==15 gather column
#define K2       360     // 10*4*9
#define K2_PAD   384
#define KT2      12      // 384/32
#define MT2      4       // 64/16 (50 padded)
#define NT2      3       // 48/16
#define K3       800     // 50*4*4
#define KT3      25      // 800/32
#define WAVES    4       // edges (or nodes) per block, one per wave32

struct Scratch {
  float    t[2 * NODE_F];                    // staged gathered input rows
  _Float16 h1[H1_SIZE + H1_PAD];             // conv1 output + zero pad
  _Float16 h2[H2_ROWS * H2_COLS + H2_PAD];   // conv2 output + slack
  float    h3[80];                           // conv3 output (75 used)
  float    hv[68];                           // hidden h[64] + edge_attr at [64]
};

// -------------------- weight fragment pre-packing --------------------------
// A-matrix 16x32 f16 lane layout (ISA 7.12.2):
//   lane L: M = L&15; K = (L>=16 ? 8 : 0) + {0..7} in halfs 0..7,
//                         +16 in halfs 8..15
__global__ void prep_w2frag_kernel(const float* __restrict__ W2,
                                   _Float16* __restrict__ out) {
  int kt = blockIdx.x >> 2;             // 0..11
  int mt = blockIdx.x & 3;              // 0..3
  int lane = threadIdx.x;               // 0..31
  int m = mt * 16 + (lane & 15);
  int klb = (lane >> 4) * 8;
  _Float16* dst = out + (size_t)(((kt * MT2 + mt) * 32) + lane) * 16;
#pragma unroll
  for (int i = 0; i < 16; ++i) {
    int k = kt * 32 + klb + (i >> 3) * 16 + (i & 7);
    float v = (m < H2_ROWS && k < K2) ? W2[m * K2 + k] : 0.f;
    dst[i] = (_Float16)v;
  }
}

__global__ void prep_w3frag_kernel(const float* __restrict__ W3,
                                   _Float16* __restrict__ out) {
  int kt = blockIdx.x;                  // 0..24
  int lane = threadIdx.x;
  int m = lane & 15;
  int klb = (lane >> 4) * 8;
  _Float16* dst = out + (size_t)((kt * 32) + lane) * 16;
#pragma unroll
  for (int i = 0; i < 16; ++i) {
    int k = kt * 32 + klb + (i >> 3) * 16 + (i & 7);
    float v = (m < 5 && k < K3) ? W3[m * K3 + k] : 0.f;
    dst[i] = (_Float16)v;
  }
}

// -------------------- shared LUT construction ------------------------------
// kLUT2[k] = h1 half-offset of im2col row k (always a valid address:
//            k >= 360 points at the zeroed pad region of h1).
// kLUT3[k] = h2 half-offset of im2col row k (all 800 rows valid).
__device__ __forceinline__ void build_luts(unsigned short* kLUT2,
                                           unsigned short* kLUT3) {
  for (int idx = threadIdx.x; idx < K2_PAD; idx += 128) {
    unsigned short v = (unsigned short)H1_SIZE;  // zero pad base
    if (idx < K2) {
      int ic = idx / 36, rem = idx % 36;
      v = (unsigned short)(ic * 144 + (rem / 9) * 16 + (rem % 9));
    }
    kLUT2[idx] = v;
  }
  for (int idx = threadIdx.x; idx < K3; idx += 128) {
    int ic = idx >> 4, rem = idx & 15;
    kLUT3[idx] = (unsigned short)(ic * H2_COLS + (rem >> 2) * 8 + (rem & 3));
  }
}

// -------------------- the conv pipeline (per wave) -------------------------
template <int IN_CH>
__device__ __forceinline__ void reduce_pipeline(
    Scratch& S, const unsigned short* kLUT2, const unsigned short* kLUT3,
    int lane,
    const float* __restrict__ W1, const float* __restrict__ b1,
    const _Float16* __restrict__ W2frag, const float* __restrict__ b2,
    const _Float16* __restrict__ W3frag, const float* __restrict__ b3,
    const float* __restrict__ Wl, const float* __restrict__ bl) {
  // ---- conv1 (VALU, tiny): pad H by 4, kernel (4,1), in_ch -> 10 ----
  for (int o = lane; o < H1_SIZE; o += 32) {
    int c = o / 144, rem = o % 144;
    int y = rem >> 4, xw = rem & 15;
    float acc = b1[c];
#pragma unroll
    for (int ky = 0; ky < 4; ++ky) {
      int hy = y + ky - 4;
      if (hy < 0 || hy > 3) continue;
#pragma unroll
      for (int ic = 0; ic < IN_CH; ++ic)
        acc += S.t[ic * NODE_F + hy * 16 + xw] * W1[(c * IN_CH + ic) * 4 + ky];
    }
    S.h1[o] = (_Float16)acc;
  }
  // zero the im2col pad region (targets of OOB LUT entries)
  for (int o = lane; o < H1_PAD; o += 32) S.h1[H1_SIZE + o] = (_Float16)0.f;
  __syncthreads();

  // ---- conv2 as WMMA GEMM: [50x360] x [360x48] ----
  int lhalf = lane & 15;
  int lhi   = lane >> 4;
  v8f zero8 = {0.f, 0.f, 0.f, 0.f, 0.f, 0.f, 0.f, 0.f};
  v8f acc2[MT2][NT2];
#pragma unroll
  for (int mt = 0; mt < MT2; ++mt)
#pragma unroll
    for (int nt = 0; nt < NT2; ++nt) acc2[mt][nt] = zero8;

  const v16h*  w2v   = (const v16h*)W2frag;
  const u16x8* lut2v = (const u16x8*)kLUT2;
  for (int kt = 0; kt < KT2; ++kt) {
    int kvec = (kt * 32 + lhi * 8) >> 3;        // 8-half units, 16B aligned
    u16x8 L0 = lut2v[kvec];
    u16x8 L1 = lut2v[kvec + 2];
    // B fragments: lane holds col n = lane&15, K rows 8h+0..7 and 16+8h+0..7
    v16h bf[NT2];
#pragma unroll
    for (int nt = 0; nt < NT2; ++nt) {
      int n = nt * 16 + lhalf;                  // spatial pos 0..47
      int noff = (n >> 3) * 16 + (n & 7);       // y*16 + xw in h1
      v16h tmp;
#pragma unroll
      for (int i = 0; i < 8; ++i) tmp[i] = S.h1[(int)L0[i] + noff];
#pragma unroll
      for (int i = 0; i < 8; ++i) tmp[8 + i] = S.h1[(int)L1[i] + noff];
      bf[nt] = tmp;
    }
#pragma unroll
    for (int mt = 0; mt < MT2; ++mt) {
      v16h af = w2v[(kt * MT2 + mt) * 32 + lane];   // pre-packed A fragment
#pragma unroll
      for (int nt = 0; nt < NT2; ++nt)
        acc2[mt][nt] = __builtin_amdgcn_wmma_f32_16x16x32_f16(
            false, af, false, bf[nt], (short)0, acc2[mt][nt], false, false);
    }
  }
  // D layout: VGPR r holds row m = r + 8*(lane>=16), col n = lane&15
#pragma unroll
  for (int mt = 0; mt < MT2; ++mt) {
#pragma unroll
    for (int r = 0; r < 8; ++r) {
      int m = mt * 16 + lhi * 8 + r;
      if (m < H2_ROWS) {
        float bb = b2[m];
#pragma unroll
        for (int nt = 0; nt < NT2; ++nt) {
          int n = nt * 16 + lhalf;
          S.h2[m * H2_COLS + n] = (_Float16)(acc2[mt][nt][r] + bb);
        }
      }
    }
  }
  __syncthreads();

  // ---- conv3 as WMMA GEMM: [5x800] x [800x15] (padded to 16x800x16) ----
  // Lanes with lhalf==15 gather garbage into B column 15; that column of D is
  // never stored, and WMMA dot products are independent per (m,n).
  v8f acc3 = zero8;
  const v16h*  w3v   = (const v16h*)W3frag;
  const u16x8* lut3v = (const u16x8*)kLUT3;
  int noff3 = (lhalf / 5) * 8 + (lhalf % 5);    // yy*8 + xx (lane 15 -> 24)
  for (int kt = 0; kt < KT3; ++kt) {
    int kvec = (kt * 32 + lhi * 8) >> 3;
    u16x8 L0 = lut3v[kvec];
    u16x8 L1 = lut3v[kvec + 2];
    v16h tmp;
#pragma unroll
    for (int i = 0; i < 8; ++i) tmp[i] = S.h2[(int)L0[i] + noff3];
#pragma unroll
    for (int i = 0; i < 8; ++i) tmp[8 + i] = S.h2[(int)L1[i] + noff3];
    v16h af = w3v[kt * 32 + lane];
    acc3 = __builtin_amdgcn_wmma_f32_16x16x32_f16(
        false, af, false, tmp, (short)0, acc3, false, false);
  }
#pragma unroll
  for (int r = 0; r < 8; ++r) {
    int m = lhi * 8 + r;
    if (m < 5 && lhalf < 15) S.h3[m * 15 + lhalf] = acc3[r] + b3[m];
  }
  __syncthreads();

  // ---- linear 75 -> 64 (VALU, tiny) ----
  for (int o = lane; o < HID; o += 32) {
    float a = bl[o];
    for (int t = 0; t < 75; ++t) a += S.h3[t] * Wl[t * HID + o];
    S.hv[o] = a;
  }
}

// -------------------- edge MPNN kernel -------------------------------------
__global__ void __launch_bounds__(128)
mpnn_edge_kernel(const float* __restrict__ xin, const float* __restrict__ ea,
                 const int* __restrict__ iidx, const int* __restrict__ jidx,
                 const float* W1, const float* b1,
                 const _Float16* W2frag, const float* b2,
                 const _Float16* W3frag, const float* b3,
                 const float* Wl, const float* bl,
                 const float* Wn, const float* bn,
                 float* __restrict__ yout, int nedges) {
  __shared__ Scratch S[WAVES];
  __shared__ alignas(16) unsigned short kLUT2[K2_PAD];
  __shared__ alignas(16) unsigned short kLUT3[K3];

  int wave = threadIdx.x >> 5;
  int lane = threadIdx.x & 31;
  Scratch& W = S[wave];
  int e = blockIdx.x * WAVES + wave;
  bool active = (e < nedges);
  int esafe = active ? e : 0;

  build_luts(kLUT2, kLUT3);

  int vi = iidx[esafe], vj = jidx[esafe];
  if (active) {
    for (int o = lane; o < 2 * NODE_F; o += 32) {
      int ch = o >> 6, off = o & 63;
      W.t[o] = xin[(size_t)(ch ? vj : vi) * NODE_F + off];
    }
  }
  __syncthreads();

  reduce_pipeline<2>(W, kLUT2, kLUT3, lane, W1, b1, W2frag, b2, W3frag, b3,
                     Wl, bl);
  if (lane == 0) W.hv[64] = ea[esafe];
  __syncthreads();

  // msg[p][b] = bn + sum_h hv[h] * Wn[p][h][b]; scatter-add at node i
  if (active) {
    for (int j = lane; j < NODE_F; j += 32) {
      int p = j >> 4, b = j & 15;
      float a = bn[j];
      for (int h = 0; h < 65; ++h) a += W.hv[h] * Wn[(p * 65 + h) * 16 + b];
      atomicAdd(&yout[(size_t)vi * NODE_F + j], a);
    }
  }
}

// -------------------- node reduce + graph readout --------------------------
__global__ void __launch_bounds__(128)
node_reduce_kernel(const float* __restrict__ xin,
                   const float* W1, const float* b1,
                   const _Float16* W2frag, const float* b2,
                   const _Float16* W3frag, const float* b3,
                   const float* Wl, const float* bl,
                   const float* __restrict__ Wg, const float* __restrict__ bg,
                   const float* __restrict__ Wf, const float* __restrict__ bf,
                   float* __restrict__ ghbar, int nnodes) {
  __shared__ Scratch S[WAVES];
  __shared__ alignas(16) unsigned short kLUT2[K2_PAD];
  __shared__ alignas(16) unsigned short kLUT3[K3];

  int wave = threadIdx.x >> 5;
  int lane = threadIdx.x & 31;
  Scratch& W = S[wave];
  int v = blockIdx.x * WAVES + wave;
  bool active = (v < nnodes);
  int vsafe = active ? v : 0;

  build_luts(kLUT2, kLUT3);

  if (active)
    for (int o = lane; o < NODE_F; o += 32)
      W.t[o] = xin[(size_t)vsafe * NODE_F + o];
  __syncthreads();

  reduce_pipeline<1>(W, kLUT2, kLUT3, lane, W1, b1, W2frag, b2, W3frag, b3,
                     Wl, bl);
  __syncthreads();

  if (active) {
    for (int d = lane; d < GDIM; d += 32) {
      float hg = bg[d], hf = bf[d];
      for (int h = 0; h < HID; ++h) {
        float hh = W.hv[h];
        hg += hh * Wg[h * GDIM + d];
        hf += hh * Wf[h * GDIM + d];
      }
      float g = 1.f / (1.f + expf(-hg));
      atomicAdd(&ghbar[d], g * hf);
    }
  }
}

// -------------------- final policy/value combine ---------------------------
__global__ void __launch_bounds__(64)
combine_kernel(const float* __restrict__ ghbar,  // [0..49]=av, [50..99]=pol
               const int* __restrict__ kptr,
               const float* __restrict__ Wp, const float* __restrict__ bp,
               const float* __restrict__ Wa, const float* __restrict__ ba,
               float* __restrict__ out) {
  __shared__ float part[64];
  int j = threadIdx.x;  // 0..63 -> (p,b)
  float kf = (float)kptr[0];
  int p = j >> 4, b = j & 15;
  float z = bp[j];
  for (int h = 0; h < GDIM; ++h) z += ghbar[GDIM + h] * Wp[(p * 51 + h) * 16 + b];
  z += kf * Wp[(p * 51 + 50) * 16 + b];
  float t = 0.1f * z;
  float xk = (t > 30.f) ? z : 10.f * logf(1.f + expf(t));  // softplus beta=0.1
  float acc = xk * Wa[51 + j];
  if (j < GDIM) acc += ghbar[j] * Wa[j];
  if (j == 0) acc += kf * Wa[50] + ba[0];
  part[j] = acc;
  __syncthreads();
  for (int s = 32; s > 0; s >>= 1) {
    if (j < s) part[j] += part[j + s];
    __syncthreads();
  }
  if (j == 0) out[0] = part[0];
}

// -------------------- host side --------------------------------------------
struct RedP { const float *W1, *W2, *W3, *Wl, *b1, *b2, *b3, *bl; };
struct MpnnP { const float *Wn, *bn; RedP red; };
struct GrepP { const float *Wf, *Wg, *bf, *bg; MpnnP bwd, fwd; RedP red; };

static RedP parse_red(void* const* d, int base) {
  RedP r;
  r.W1 = (const float*)d[base + 0]; r.W2 = (const float*)d[base + 1];
  r.W3 = (const float*)d[base + 2]; r.Wl = (const float*)d[base + 3];
  r.b1 = (const float*)d[base + 4]; r.b2 = (const float*)d[base + 5];
  r.b3 = (const float*)d[base + 6]; r.bl = (const float*)d[base + 7];
  return r;
}
static MpnnP parse_mpnn(void* const* d, int base) {
  MpnnP m;
  m.Wn = (const float*)d[base + 0];
  m.bn = (const float*)d[base + 1];
  m.red = parse_red(d, base + 2);
  return m;
}
static GrepP parse_grep(void* const* d, int base) {  // 32 leaves, sorted keys
  GrepP g;
  g.Wf = (const float*)d[base + 0]; g.Wg = (const float*)d[base + 1];
  g.bf = (const float*)d[base + 2]; g.bg = (const float*)d[base + 3];
  g.bwd = parse_mpnn(d, base + 4);
  g.fwd = parse_mpnn(d, base + 14);
  g.red = parse_red(d, base + 24);
  return g;
}

extern "C" void kernel_launch(void* const* d_in, const int* in_sizes, int n_in,
                              void* d_out, int out_size, void* d_ws,
                              size_t ws_size, hipStream_t stream) {
  const float* x    = (const float*)d_in[0];
  const float* ea   = (const float*)d_in[1];
  const int*   eidx = (const int*)d_in[2];
  const int*   kptr = (const int*)d_in[3];
  const float* Wa   = (const float*)d_in[4];
  const float* Wp   = (const float*)d_in[5];
  GrepP av = parse_grep(d_in, 6);
  const float* ba = (const float*)d_in[38];
  const float* bp = (const float*)d_in[39];
  GrepP pol = parse_grep(d_in, 40);

  const int E = in_sizes[1];
  const int V = in_sizes[0] / NODE_F;
  const int* src = eidx;
  const int* dst = eidx + E;

  // workspace carving (256B aligned)
  char* wsb = (char*)d_ws;
  size_t off = 0;
  auto carve = [&](size_t bytes) -> void* {
    void* p = wsb + off;
    off = (off + bytes + 255) & ~(size_t)255;
    return p;
  };
  _Float16* w2f[6];
  _Float16* w3f[6];
  for (int s = 0; s < 6; ++s) w2f[s] = (_Float16*)carve((size_t)KT2 * MT2 * 32 * 16 * 2);
  for (int s = 0; s < 6; ++s) w3f[s] = (_Float16*)carve((size_t)KT3 * 32 * 16 * 2);
  float* xbufA = (float*)carve((size_t)V * NODE_F * 4);
  float* xbufB = (float*)carve((size_t)V * NODE_F * 4);
  float* ghbar = (float*)carve(2 * GDIM * 4);
  (void)ws_size; (void)n_in; (void)out_size;

  // sets: 0=av.bwd 1=av.fwd 2=av.node 3=pol.bwd 4=pol.fwd 5=pol.node
  const RedP* sets[6] = {&av.bwd.red, &av.fwd.red, &av.red,
                         &pol.bwd.red, &pol.fwd.red, &pol.red};
  for (int s = 0; s < 6; ++s) {
    prep_w2frag_kernel<<<KT2 * MT2, 32, 0, stream>>>(sets[s]->W2, w2f[s]);
    prep_w3frag_kernel<<<KT3, 32, 0, stream>>>(sets[s]->W3, w3f[s]);
  }
  hipMemsetAsync(ghbar, 0, 2 * GDIM * 4, stream);

  int egrid = (E + WAVES - 1) / WAVES;
  int vgrid = (V + WAVES - 1) / WAVES;

  for (int r = 0; r < 2; ++r) {
    const GrepP& G = r ? pol : av;
    const float* xcur = x;
    for (int step = 0; step < 2; ++step) {
      float* xn = step ? xbufB : xbufA;
      hipMemsetAsync(xn, 0, (size_t)V * NODE_F * 4, stream);
      // fwd: j=src, i=dst (aggregate at dst)
      mpnn_edge_kernel<<<egrid, 128, 0, stream>>>(
          xcur, ea, dst, src,
          G.fwd.red.W1, G.fwd.red.b1, w2f[r * 3 + 1], G.fwd.red.b2,
          w3f[r * 3 + 1], G.fwd.red.b3, G.fwd.red.Wl, G.fwd.red.bl,
          G.fwd.Wn, G.fwd.bn, xn, E);
      // bwd: j=dst, i=src (aggregate at src)
      mpnn_edge_kernel<<<egrid, 128, 0, stream>>>(
          xcur, ea, src, dst,
          G.bwd.red.W1, G.bwd.red.b1, w2f[r * 3 + 0], G.bwd.red.b2,
          w3f[r * 3 + 0], G.bwd.red.b3, G.bwd.red.Wl, G.bwd.red.bl,
          G.bwd.Wn, G.bwd.bn, xn, E);
      xcur = xn;
    }
    node_reduce_kernel<<<vgrid, 128, 0, stream>>>(
        xcur,
        G.red.W1, G.red.b1, w2f[r * 3 + 2], G.red.b2, w3f[r * 3 + 2],
        G.red.b3, G.red.Wl, G.red.bl,
        G.Wg, G.bg, G.Wf, G.bf, ghbar + r * GDIM, V);
  }

  combine_kernel<<<1, 64, 0, stream>>>(ghbar, kptr, Wp, bp, Wa, ba,
                                       (float*)d_out);
}
